// THNN_ab_90185723281667
// MI455X (gfx1250) — matrix operand order repaired and checked by reading, not compiled, not run on hardware
//
#include <hip/hip_runtime.h>
#include <hip/hip_bf16.h>
#include <math.h>

typedef __attribute__((ext_vector_type(16))) _Float16 v16h;
typedef __attribute__((ext_vector_type(8)))  float    v8f;

#define DEV_INLINE __device__ __forceinline__

// ---------------------------------------------------------------------------
// WMMA helper (CDNA5 wave32, 16x16x32 f16 -> f32)
// ---------------------------------------------------------------------------
DEV_INLINE v8f wmma_f16(v16h a, v16h b, v8f c) {
  return __builtin_amdgcn_wmma_f32_16x16x32_f16(false, a, false, b, (short)0, c,
                                                false, false);
}

// Packed-layout fragment loads: fragments are stored so that each lane reads
// 16 contiguous f16 (32 bytes).
//
// A layout (16x32, ISA 7.12.2): lane group g holds slots {8g..8g+7, 16+8g..23+8g}.
// Inverse map for packing:  g = (slot>>3)&1 ; i = (slot&7) + 8*(slot>>4).
// Packed index: (((kc*2+g)*16 + m)*16 + i)
DEV_INLINE v16h load_a_packed(const _Float16* Ls, int m, int g, int kc) {
  return *(const v16h*)(Ls + ((((kc * 2 + g) * 16) + m) << 4));
}

// B layout (32x16): lane group g holds K = kc*32 + {8g..8g+7, 16+8g..23+8g}, same order as A.
// Packed index: (((kc*2+g)*ncols + col)*16 + i)
DEV_INLINE v16h load_b_packed(const _Float16* P, int ncols, int col, int g,
                              int kc) {
  return *(const v16h*)(P + ((((kc * 2 + g) * ncols) + col) << 4));
}

// ---------------------------------------------------------------------------
// Weight pre-pack: f32 row-major W[K][ncols] -> f16 WMMA-B fragment order
// ---------------------------------------------------------------------------
__global__ void pack_w_kernel(const float* __restrict__ W,
                              _Float16* __restrict__ P, int total, int ncols) {
  int idx = blockIdx.x * blockDim.x + threadIdx.x;
  if (idx < total) {
    int k = idx / ncols, col = idx - k * ncols;
    int kc = k >> 5, rem = k & 31, g = (rem >> 3) & 1, i = (rem & 7) + 8 * (rem >> 4);
    P[((((kc * 2 + g) * ncols) + col) << 4) + i] = (_Float16)W[idx];
  }
}

// ---------------------------------------------------------------------------
// Small utility kernels
// ---------------------------------------------------------------------------
__global__ void zero_kernel(float* __restrict__ out, float* __restrict__ deg,
                            int nOut, int nDeg) {
  int i = blockIdx.x * blockDim.x + threadIdx.x;
  int stride = gridDim.x * blockDim.x;
  for (int idx = i; idx < nOut; idx += stride) out[idx] = 0.0f;
  for (int idx = i; idx < nDeg; idx += stride) deg[idx] = 0.0f;
}

__global__ void deg_kernel(const int* __restrict__ edges, float* __restrict__ deg,
                           int total) {
  int i = blockIdx.x * blockDim.x + threadIdx.x;
  if (i < total) atomicAdd(&deg[edges[i]], 1.0f);
}

__global__ void finalize_kernel(float* __restrict__ out,
                                const float* __restrict__ deg, int total) {
  int i = blockIdx.x * blockDim.x + threadIdx.x;
  if (i < total) {
    float d = deg[i >> 7];  // OUT = 128
    d = d > 1.0f ? d : 1.0f;
    float v = out[i] / d;
    out[i] = v > 0.0f ? v : 0.0f;
  }
}

// ---------------------------------------------------------------------------
// Fused node MLP: emb_p = X@Wp+bp ; H = relu(X@W1+b1) (LDS only) ;
// emb_2 = H@W2+b2.  16 node rows per block, 8 waves, WMMA f16->f32.
// LDS tiles are stored directly in A-fragment order.
// ---------------------------------------------------------------------------
__global__ __launch_bounds__(256) void node_mlp_kernel(
    const float* __restrict__ X,
    const _Float16* __restrict__ Wpp, const float* __restrict__ bp,
    const _Float16* __restrict__ W1p, const float* __restrict__ b1,
    const _Float16* __restrict__ W2p, const float* __restrict__ b2,
    float* __restrict__ emb_p, float* __restrict__ emb_2) {
  __shared__ __align__(32) _Float16 Xs[16 * 256];  // 8 KB, A-frag order
  __shared__ __align__(32) _Float16 Hs[16 * 512];  // 16 KB, A-frag order

  const int tid = threadIdx.x;
  const int wave = tid >> 5, lane = tid & 31;
  const int g = lane >> 4, n = lane & 15, m = lane & 15;
  const int rowBase = blockIdx.x * 16;

  // Stage X tile into LDS as f16, directly in A-fragment order
  for (int idx = tid; idx < 16 * 256; idx += 256) {
    int r = idx >> 8, c = idx & 255;
    int kc = c >> 5, slot = c & 31;
    int g2 = (slot >> 3) & 1;
    int i2 = (slot & 7) + 8 * (slot >> 4);
    Xs[((((kc * 2 + g2) * 16) + r) << 4) + i2] =
        (_Float16)X[(rowBase + r) * 256 + c];
  }
  __syncthreads();

  // ---- H = relu(X@W1 + b1): 32 col tiles, 4 per wave ----
  v8f accH[4];
#pragma unroll
  for (int t = 0; t < 4; ++t) { v8f z = {}; accH[t] = z; }
  for (int kc = 0; kc < 8; ++kc) {
    v16h a = load_a_packed(Xs, m, g, kc);
#pragma unroll
    for (int t = 0; t < 4; ++t) {
      v16h b = load_b_packed(W1p, 512, (wave * 4 + t) * 16 + n, g, kc);
      accH[t] = wmma_f16(a, b, accH[t]);
    }
  }

  // ---- emb_p = X@Wp + bp: 4 col tiles, waves 0..3 (uniform per wave) ----
  if (wave < 4) {
    v8f accP = {};
    for (int kc = 0; kc < 8; ++kc) {
      v16h a = load_a_packed(Xs, m, g, kc);
      v16h b = load_b_packed(Wpp, 64, wave * 16 + n, g, kc);
      accP = wmma_f16(a, b, accP);
    }
    float bias = bp[wave * 16 + n];
#pragma unroll
    for (int v = 0; v < 8; ++v) {
      int r = rowBase + v + 8 * g;  // C/D layout: M = v + 8*(lane>=16)
      emb_p[r * 64 + wave * 16 + n] = accP[v] + bias;
    }
  }

  // ---- write H tile (bias + relu) into LDS in A-fragment order ----
#pragma unroll
  for (int t = 0; t < 4; ++t) {
    int ct = wave * 4 + t;
    float bias = b1[ct * 16 + n];
    int col = ct * 16 + n;
    int kc2 = col >> 5, slot = col & 31;
    int g2 = (slot >> 3) & 1;
    int i2 = (slot & 7) + 8 * (slot >> 4);
#pragma unroll
    for (int v = 0; v < 8; ++v) {
      float hv = accH[t][v] + bias;
      hv = hv > 0.0f ? hv : 0.0f;
      Hs[((((kc2 * 2 + g2) * 16) + (v + 8 * g)) << 4) + i2] = (_Float16)hv;
    }
  }
  __syncthreads();

  // ---- emb_2 = H@W2 + b2: 8 col tiles, one per wave ----
  v8f acc2 = {};
  for (int kc = 0; kc < 16; ++kc) {
    v16h a = load_a_packed(Hs, m, g, kc);
    v16h b = load_b_packed(W2p, 128, wave * 16 + n, g, kc);
    acc2 = wmma_f16(a, b, acc2);
  }
  float bias2 = b2[wave * 16 + n];
#pragma unroll
  for (int v = 0; v < 8; ++v) {
    int r = rowBase + v + 8 * g;
    emb_2[r * 128 + wave * 16 + n] = acc2[v] + bias2;
  }
}

// ---------------------------------------------------------------------------
// Edge message kernel: 4 edges per wave -> 16 A-rows (edge,member).
// g = emb_p[node]*deg^(1/4); exclusive product over the 4 members via
// shfl_xor; t = tanh(scale*excl/6); msg = t@Wq + bq + relu(sum emb_2);
// atomic scatter into out accumulator.
// A lane's K slots are two runs of 8 consecutive values -> float4 gathers.
// ---------------------------------------------------------------------------
__global__ __launch_bounds__(256) void edge_msg_kernel(
    const int* __restrict__ edges,
    const float* __restrict__ emb_p, const float* __restrict__ emb_2,
    const float* __restrict__ deg,
    const _Float16* __restrict__ Wqp, const float* __restrict__ bq,
    float* __restrict__ outAcc) {
  const int tid = threadIdx.x;
  const int wave = tid >> 5, lane = tid & 31;
  const int g = lane >> 4, n = lane & 15, m = lane & 15;
  const int e0 = (blockIdx.x * 8 + wave) * 4;  // first of 4 edges for this wave

  // This lane's A-matrix row: edge = e0 + (m>>2), member = m&3
  const int eRow = e0 + (m >> 2);
  const int node = edges[eRow * 4 + (m & 3)];
  const float sc = powf(deg[node], 0.25f);
  const float* ep = emb_p + node * 64;

  // C/D rows of this lane map to two edges (4 members each)
  const int eA = e0 + 2 * g, eB = eA + 1;
  int nA[4], nB[4];
#pragma unroll
  for (int j = 0; j < 4; ++j) {
    nA[j] = edges[eA * 4 + j];
    nB[j] = edges[eB * 4 + j];
  }

  v8f C[8];
#pragma unroll
  for (int t = 0; t < 8; ++t) { v8f z = {}; C[t] = z; }

  const float inv6 = 1.0f / 6.0f;  // 1/(k-1)! for k=4
#pragma unroll
  for (int c = 0; c < 2; ++c) {  // RANK=64 -> two K=32 chunks
    // lane's 16 K slots = ep[32c+8g .. +7] and ep[32c+16+8g .. +7]
    const float* plo = ep + 32 * c + 8 * g;
    const float* phi = ep + 32 * c + 16 + 8 * g;
    float4 l0 = *(const float4*)(plo);
    float4 l1 = *(const float4*)(plo + 4);
    float4 h0 = *(const float4*)(phi);
    float4 h1 = *(const float4*)(phi + 4);
    float gk[16] = {l0.x, l0.y, l0.z, l0.w, l1.x, l1.y, l1.z, l1.w,
                    h0.x, h0.y, h0.z, h0.w, h1.x, h1.y, h1.z, h1.w};
    v16h a;
#pragma unroll
    for (int i = 0; i < 16; ++i) {
      float gv = gk[i] * sc;
      // exclusive product over the aligned 4-lane member cluster
      float a1 = __shfl_xor(gv, 1);  // partner within pair
      float pp = gv * a1;            // pair product
      float op = __shfl_xor(pp, 2);  // other pair's product
      float excl = a1 * op;          // product of the 3 other members
      a[i] = (_Float16)tanhf(inv6 * sc * excl);
    }
#pragma unroll
    for (int nt = 0; nt < 8; ++nt) {  // OUT=128 -> 8 N-tiles
      v16h b = load_b_packed(Wqp, 128, nt * 16 + n, g, c);
      C[nt] = wmma_f16(a, b, C[nt]);
    }
  }

  // Epilogue: + bq + relu(edge_sum2), atomic scatter-add per member node
#pragma unroll
  for (int nt = 0; nt < 8; ++nt) {
    int col = nt * 16 + n;
    float bqv = bq[col];
    float sA = 0.0f, sB = 0.0f;
#pragma unroll
    for (int j = 0; j < 4; ++j) {
      sA += emb_2[nA[j] * 128 + col];
      sB += emb_2[nB[j] * 128 + col];
    }
    sA = sA > 0.0f ? sA : 0.0f;
    sB = sB > 0.0f ? sB : 0.0f;
#pragma unroll
    for (int v = 0; v < 8; ++v) {
      int nd = (v < 4) ? nA[v & 3] : nB[v & 3];
      float es = (v < 4) ? sA : sB;
      atomicAdd(&outAcc[nd * 128 + col], C[nt][v] + bqv + es);
    }
  }
}

// ---------------------------------------------------------------------------
// Launch
// ---------------------------------------------------------------------------
extern "C" void kernel_launch(void* const* d_in, const int* in_sizes, int n_in,
                              void* d_out, int out_size, void* d_ws, size_t ws_size,
                              hipStream_t stream) {
  const float* X     = (const float*)d_in[0];
  const int*   edges = (const int*)d_in[1];
  const float* Wp    = (const float*)d_in[2];
  const float* bp    = (const float*)d_in[3];
  const float* Wq    = (const float*)d_in[4];
  const float* bq    = (const float*)d_in[5];
  const float* W1    = (const float*)d_in[6];
  const float* b1    = (const float*)d_in[7];
  const float* W2    = (const float*)d_in[8];
  const float* b2    = (const float*)d_in[9];
  float* out = (float*)d_out;

  const int FEAT = 256, HID = 512, OUT = 128, RANK = 64, K = 4;
  const int N = in_sizes[0] / FEAT;  // 50000 (divisible by 16)
  const int E = in_sizes[1] / K;     // 100000 (divisible by 32)

  // Workspace layout (f32 region, then 32B-aligned f16 packed-weight region)
  float* ws    = (float*)d_ws;
  float* emb_p = ws;                        // N*RANK floats
  float* emb_2 = emb_p + (size_t)N * RANK;  // N*OUT floats
  float* deg   = emb_2 + (size_t)N * OUT;   // N floats
  _Float16* h  = (_Float16*)(deg + N);      // offset 38,600,000 B (32B aligned)
  _Float16* W1p = h;                        // FEAT*HID halves
  _Float16* W2p = W1p + FEAT * HID;         // HID*OUT halves
  _Float16* Wpp = W2p + HID * OUT;          // FEAT*RANK halves
  _Float16* Wqp = Wpp + FEAT * RANK;        // RANK*OUT halves

  zero_kernel<<<4096, 256, 0, stream>>>(out, deg, N * OUT, N);
  pack_w_kernel<<<(FEAT * HID + 255) / 256, 256, 0, stream>>>(W1, W1p,
                                                              FEAT * HID, HID);
  pack_w_kernel<<<(HID * OUT + 255) / 256, 256, 0, stream>>>(W2, W2p,
                                                             HID * OUT, OUT);
  pack_w_kernel<<<(FEAT * RANK + 255) / 256, 256, 0, stream>>>(Wp, Wpp,
                                                               FEAT * RANK, RANK);
  pack_w_kernel<<<(RANK * OUT + 255) / 256, 256, 0, stream>>>(Wq, Wqp,
                                                              RANK * OUT, OUT);
  deg_kernel<<<(E * K + 255) / 256, 256, 0, stream>>>(edges, deg, E * K);
  node_mlp_kernel<<<N / 16, 256, 0, stream>>>(X, Wpp, bp, W1p, b1, W2p, b2,
                                              emb_p, emb_2);
  edge_msg_kernel<<<E / 32, 256, 0, stream>>>(edges, emb_p, emb_2, deg, Wqp, bq,
                                              out);
  finalize_kernel<<<(N * OUT + 255) / 256, 256, 0, stream>>>(out, deg, N * OUT);
}
